// LSTM_34943853920913
// MI455X (gfx1250) — matrix operand order repaired and checked by reading, not compiled, not run on hardware
//
#include <hip/hip_runtime.h>
#include <hip/hip_bf16.h>

// ---------------------------------------------------------------------------
// LSTM on MI455X (gfx1250), wave32 + WMMA bf16.
//
// Sizes: L=512, N=64, D_IN=512, D_H=512, gates K = 1024, gate cols = 2048.
//
//  * Persistent kernel: 32 workgroups x 256 threads. WG w owns hidden slice
//    [16w,16w+16) across all 4 gates (64 gate columns) -> cell state c stays
//    in LDS; only h (bf16) crosses the one grid barrier per timestep.
//  * WG's gate-weight slice (128 KB bf16, pre-swizzled into per-lane WMMA
//    B-fragment order) is copied to LDS ONCE; hot loop = 2 global b128 (A)
//    + 4 ds_load_b128 (B) + 2 v_wmma per k-tile. K loop split into x-half /
//    h-half so all addresses are constant offsets.
//  * Next step's x-fragments are prefetched (global_prefetch_b8) before the
//    grid barrier: hides post-barrier load latency on the serial chain.
//  * h history stored bf16 (slot t+1 = h_t); reused as A operand for the next
//    step and for the final (L*N,512)@(512,512) WMMA projection.
//
// Workspace (~69 MB): [0,256) barrier | xbf 32MB | hsb 32MB | wg 4MB | wout 0.5MB
// LDS per WG: 128K weights + 16K z + 4K c = 148 KB (<320 KB WGP LDS).
// ---------------------------------------------------------------------------

typedef __attribute__((ext_vector_type(16))) __bf16 v16bf;
typedef __attribute__((ext_vector_type(8)))  float  v8f;
typedef __attribute__((ext_vector_type(4)))  unsigned int v4u;

union FragB16 { v16bf bf; v4u q[2]; };

__device__ inline v8f wmma_bf16(const FragB16& a, const FragB16& b, v8f c) {
    // (neg_a, A, neg_b, B, c_mod, C, reuse_a, reuse_b)
    return __builtin_amdgcn_wmma_f32_16x16x32_bf16(false, a.bf, false, b.bf,
                                                   (short)0, c, false, false);
}

__device__ inline unsigned short f2bf(float f) {
    unsigned u = __float_as_uint(f);
    unsigned r = (u + 0x7FFFu + ((u >> 16) & 1u)) >> 16;  // RNE
    return (unsigned short)r;
}

__device__ inline float sigmoid_(float x) { return 1.0f / (1.0f + __expf(-x)); }
__device__ inline float tanh_(float x) {
    float e = __expf(2.0f * x);
    return 1.0f - 2.0f / (e + 1.0f);   // correct limits at +-inf
}

__device__ inline void grid_sync(unsigned* cnt, unsigned* gen, unsigned nb) {
    __threadfence();            // publish this block's global stores (agent scope)
    __syncthreads();
    if (threadIdx.x == 0) {
        unsigned g = __hip_atomic_load(gen, __ATOMIC_RELAXED, __HIP_MEMORY_SCOPE_AGENT);
        unsigned p = __hip_atomic_fetch_add(cnt, 1u, __ATOMIC_ACQ_REL, __HIP_MEMORY_SCOPE_AGENT);
        if (p == nb - 1u) {
            __hip_atomic_store(cnt, 0u, __ATOMIC_RELAXED, __HIP_MEMORY_SCOPE_AGENT);
            __hip_atomic_fetch_add(gen, 1u, __ATOMIC_RELEASE, __HIP_MEMORY_SCOPE_AGENT);
        } else {
            while (__hip_atomic_load(gen, __ATOMIC_ACQUIRE, __HIP_MEMORY_SCOPE_AGENT) == g) {
                __builtin_amdgcn_s_sleep(1);
            }
        }
    }
    __syncthreads();
}

// -------------------------- setup kernels ----------------------------------

__global__ __launch_bounds__(256) void cvt_f32_bf16_kernel(
    const float* __restrict__ src, unsigned short* __restrict__ dst, unsigned n) {
    unsigned i = blockIdx.x * 256u + threadIdx.x;
    if (i < n) dst[i] = f2bf(src[i]);
}

// Swizzle 4 gate weight matrices (each (1024,512) f32) into WMMA-B fragment
// order, gate-interleaved per hidden-slice:
//   tile = w*4 + g ; frag(tile,kt) = 512 bf16, lane holds 16 contiguous K
//   values of column col = g*512 + w*16 + (lane&15); K = kt*32 + hi*16 + i.
__global__ __launch_bounds__(256) void swz_gates_kernel(
    const float* __restrict__ Wf, const float* __restrict__ Wc,
    const float* __restrict__ Wi, const float* __restrict__ Wo,
    unsigned short* __restrict__ dst) {
    unsigned idx = blockIdx.x * 256u + threadIdx.x;          // 0 .. 2M-1
    unsigned i    = idx & 15u;
    unsigned lane = (idx >> 4) & 31u;
    unsigned frag = idx >> 9;                                // tile*32 + kt
    unsigned kt   = frag & 31u;
    unsigned tile = frag >> 5;                               // 0..127
    unsigned g    = tile & 3u;
    unsigned w    = tile >> 2;
    unsigned k    = kt * 32u + ((lane >> 4) ? 16u : 0u) + i; // 0..1023
    unsigned col  = w * 16u + (lane & 15u);                  // 0..511 within gate
    const float* W = (g == 0u) ? Wf : (g == 1u) ? Wc : (g == 2u) ? Wi : Wo;
    dst[idx] = f2bf(W[(size_t)k * 512u + col]);
}

// Same fragment order for W_out (512,512): frag index = ntile*16 + kt.
__global__ __launch_bounds__(256) void swz_wout_kernel(
    const float* __restrict__ Wout, unsigned short* __restrict__ dst) {
    unsigned idx = blockIdx.x * 256u + threadIdx.x;          // 0 .. 262143
    unsigned i    = idx & 15u;
    unsigned lane = (idx >> 4) & 31u;
    unsigned frag = idx >> 9;                                // ntile*16 + kt
    unsigned kt   = frag & 15u;
    unsigned nt   = frag >> 4;                               // 0..31
    unsigned k    = kt * 32u + ((lane >> 4) ? 16u : 0u) + i; // 0..511
    unsigned col  = nt * 16u + (lane & 15u);                 // 0..511
    dst[idx] = f2bf(Wout[(size_t)k * 512u + col]);
}

// -------------------------- persistent LSTM --------------------------------

__global__ __launch_bounds__(256) void lstm_persistent_kernel(
    const float* __restrict__ c0,
    const float* __restrict__ b_f, const float* __restrict__ b_c,
    const float* __restrict__ b_i, const float* __restrict__ b_o,
    const unsigned short* __restrict__ xbf,   // (L,64,512) bf16
    unsigned short* __restrict__ hsb,         // (L+1,64,512) bf16, slot0=h0
    const unsigned short* __restrict__ wg,    // swizzled gate weights
    float* __restrict__ out_h, float* __restrict__ out_c,
    unsigned* __restrict__ bar) {
    __shared__ unsigned short wls[4 * 32 * 512];  // 128 KB: WG's weight slice
    __shared__ float zls[64 * 64];                //  16 KB: z slice
    __shared__ float cls[64 * 16];                //   4 KB: cell state (fp32)

    const int w    = blockIdx.x;          // hidden slice [16w,16w+16)
    const int tid  = threadIdx.x;
    const int lane = tid & 31;
    const int wid  = tid >> 5;            // 0..7
    const int mtile = wid & 3;            // batch tile (16 rows)
    const int gpair = wid >> 2;           // gate pair: {0,1} or {2,3}
    const int g0 = gpair * 2, g1 = g0 + 1;

    // per-lane fragment geometry (this code's contiguous k = 16*hi + s, same order as B)
    const int mrow = mtile * 16 + (lane & 15);
    const int kbA  = (lane >> 4) ? 16 : 0;

    // warm caches for the very first x row (otherwise cold HBM miss at t=0)
    {
        const unsigned short* nx = xbf + (size_t)mrow * 512 + kbA;
#pragma unroll
        for (int j = 0; j < 8; ++j) __builtin_prefetch(nx + j * 64, 0, 3);
    }
    // one-time: weight slice -> LDS (4 tiles x 32 ktiles x 1KB, b128 copies)
    {
        const v4u* src = (const v4u*)(wg + (size_t)w * (4u * 32u * 512u));
        v4u* dst = (v4u*)wls;
        for (int i = tid; i < 8192; i += 256) dst[i] = src[i];
    }
    // preload cell state
    for (int e = tid; e < 1024; e += 256) {
        int n = e >> 4, j = e & 15;
        cls[e] = c0[n * 512 + w * 16 + j];
    }
    __syncthreads();

    const unsigned short* bb0 = wls + g0 * (32 * 512) + lane * 16;
    const unsigned short* bb1 = wls + g1 * (32 * 512) + lane * 16;

    for (int t = 0; t < 512; ++t) {
        // ---- fused-gate GEMM: z[64,64] = [x_t ; h_{t-1}] @ Wg_slice ----
        v8f acc0 = {}, acc1 = {};
        const unsigned short* arow_x = xbf + ((size_t)t * 64 + mrow) * 512 + kbA;
        const unsigned short* arow_h = hsb + ((size_t)t * 64 + mrow) * 512 + kbA;
#pragma unroll 4
        for (int kt = 0; kt < 16; ++kt) {          // K 0..511 : x part
            FragB16 a, b0, b1;
            const unsigned short* ap = arow_x + kt * 32;
            a.q[0] = *(const v4u*)(ap);
            a.q[1] = *(const v4u*)(ap + 8);
            const unsigned short* p0 = bb0 + kt * 512;
            const unsigned short* p1 = bb1 + kt * 512;
            b0.q[0] = *(const v4u*)(p0);
            b0.q[1] = *(const v4u*)(p0 + 8);
            b1.q[0] = *(const v4u*)(p1);
            b1.q[1] = *(const v4u*)(p1 + 8);
            acc0 = wmma_bf16(a, b0, acc0);
            acc1 = wmma_bf16(a, b1, acc1);
        }
#pragma unroll 4
        for (int kt = 0; kt < 16; ++kt) {          // K 512..1023 : h part
            FragB16 a, b0, b1;
            const unsigned short* ap = arow_h + kt * 32;
            a.q[0] = *(const v4u*)(ap);
            a.q[1] = *(const v4u*)(ap + 8);
            const unsigned short* p0 = bb0 + (16 + kt) * 512;
            const unsigned short* p1 = bb1 + (16 + kt) * 512;
            b0.q[0] = *(const v4u*)(p0);
            b0.q[1] = *(const v4u*)(p0 + 8);
            b1.q[0] = *(const v4u*)(p1);
            b1.q[1] = *(const v4u*)(p1 + 8);
            acc0 = wmma_bf16(a, b0, acc0);
            acc1 = wmma_bf16(a, b1, acc1);
        }
        // D fragment -> LDS (f32 C/D layout: vgpr r = row r (+8 for hi lanes))
        const int zrow = mtile * 16 + ((lane >> 4) ? 8 : 0);
        const int n0 = lane & 15;
#pragma unroll
        for (int r = 0; r < 8; ++r) {
            zls[(zrow + r) * 64 + g0 * 16 + n0] = acc0[r];
            zls[(zrow + r) * 64 + g1 * 16 + n0] = acc1[r];
        }
        __syncthreads();

        // ---- gate math + cell update (4 hidden elements per thread) ----
#pragma unroll
        for (int i = 0; i < 4; ++i) {
            int e = tid + 256 * i;            // = n*16 + j
            int n = e >> 4, j = e & 15;
            int col = w * 16 + j;
            float zf = zls[n * 64 + j]      + b_f[col];
            float zc = zls[n * 64 + 16 + j] + b_c[col];
            float zi = zls[n * 64 + 32 + j] + b_i[col];
            float zo = zls[n * 64 + 48 + j] + b_o[col];
            float f  = sigmoid_(zf);
            float Ct = tanh_(zc);
            float it = sigmoid_(zi);
            float ot = sigmoid_(zo);
            float c  = f * cls[e] + it * Ct;
            cls[e] = c;
            float h  = ot * tanh_(c);
            hsb[((size_t)(t + 1) * 64 + n) * 512 + col] = f2bf(h);
            if (t == 511) out_h[n * 512 + col] = h;
        }

        // warm WGP$/L2 with next step's x fragments while we sit in the
        // barrier (global_prefetch_b8: no VGPR result, no counter traffic).
        if (t < 511) {
            const unsigned short* nx = arow_x + 64 * 512;   // same row, t+1
#pragma unroll
            for (int j = 0; j < 8; ++j) __builtin_prefetch(nx + j * 64, 0, 3);
        }

        grid_sync(bar, bar + 1, gridDim.x);   // h_t published; c stays in LDS
    }

    for (int e = tid; e < 1024; e += 256)
        out_c[(e >> 4) * 512 + w * 16 + (e & 15)] = cls[e];
}

// -------------------------- output projection ------------------------------
// outputs(32768,512) = hs(32768,512) @ W_out(512,512) + b_out.
// hs rows r = t*64+n map to hsb rows r+64 (slot t+1 holds h_t).
// One block per M-tile; wave wid covers 4 N-tiles -> each A fragment feeds
// 4 WMMAs.
__global__ __launch_bounds__(256) void outproj_kernel(
    const unsigned short* __restrict__ hsb,
    const unsigned short* __restrict__ wout,
    const float* __restrict__ b_out, float* __restrict__ out) {
    const int lane = threadIdx.x & 31;
    const int wid  = threadIdx.x >> 5;                  // 0..7
    const size_t mtile = blockIdx.x;                    // 0..2047
    const int nt0 = wid * 4;                            // 4 ntiles per wave

    const int kbA = (lane >> 4) ? 16 : 0;
    const size_t arow = mtile * 16 + (lane & 15) + 64;  // +64: slot shift
    const unsigned short* ap_base = hsb + arow * 512 + kbA;
    const unsigned short* bb = wout + (size_t)nt0 * (16u * 512u) + lane * 16;

    v8f acc[4] = {{}, {}, {}, {}};
#pragma unroll 2
    for (int kt = 0; kt < 16; ++kt) {
        FragB16 a;
        const unsigned short* ap = ap_base + kt * 32;
        a.q[0] = *(const v4u*)(ap);
        a.q[1] = *(const v4u*)(ap + 8);
#pragma unroll
        for (int j = 0; j < 4; ++j) {
            FragB16 b;
            const unsigned short* bp = bb + (size_t)j * (16u * 512u) + kt * 512;
            b.q[0] = *(const v4u*)(bp);
            b.q[1] = *(const v4u*)(bp + 8);
            acc[j] = wmma_bf16(a, b, acc[j]);
        }
    }
    const size_t row0 = mtile * 16 + ((lane >> 4) ? 8 : 0);
#pragma unroll
    for (int j = 0; j < 4; ++j) {
        const int col = (nt0 + j) * 16 + (lane & 15);
        const float bias = b_out[col];
#pragma unroll
        for (int r = 0; r < 8; ++r)
            out[(row0 + r) * 512 + col] = acc[j][r] + bias;
    }
}

// ---------------------------------------------------------------------------

extern "C" void kernel_launch(void* const* d_in, const int* in_sizes, int n_in,
                              void* d_out, int out_size, void* d_ws, size_t ws_size,
                              hipStream_t stream) {
    const float* x    = (const float*)d_in[0];
    const float* h0   = (const float*)d_in[1];
    const float* c0   = (const float*)d_in[2];
    const float* Wf   = (const float*)d_in[3];
    const float* bf_  = (const float*)d_in[4];
    const float* Wc   = (const float*)d_in[5];
    const float* bc_  = (const float*)d_in[6];
    const float* Wi   = (const float*)d_in[7];
    const float* bi_  = (const float*)d_in[8];
    const float* Wo   = (const float*)d_in[9];
    const float* bo_  = (const float*)d_in[10];
    const float* Wout = (const float*)d_in[11];
    const float* bout = (const float*)d_in[12];

    float* out   = (float*)d_out;                      // (512,64,512)
    float* out_h = out + (size_t)512 * 64 * 512;       // (64,512)
    float* out_c = out_h + (size_t)64 * 512;           // (64,512)

    char* ws = (char*)d_ws;
    const size_t XBF_ELEMS  = (size_t)512 * 64 * 512;         // 16,777,216
    const size_t HSB_ELEMS  = (size_t)513 * 64 * 512;         // 16,809,984
    const size_t WG_ELEMS   = (size_t)1024 * 2048;            //  2,097,152
    unsigned*       bar   = (unsigned*)ws;
    unsigned short* xbf   = (unsigned short*)(ws + 256);
    unsigned short* hsb   = xbf + XBF_ELEMS;
    unsigned short* wg    = hsb + HSB_ELEMS;
    unsigned short* wouts = wg + WG_ELEMS;

    hipMemsetAsync(bar, 0, 256, stream);   // barrier counters (capture-safe)

    cvt_f32_bf16_kernel<<<(unsigned)(XBF_ELEMS / 256), 256, 0, stream>>>(x, xbf, (unsigned)XBF_ELEMS);
    cvt_f32_bf16_kernel<<<128, 256, 0, stream>>>(h0, hsb, 64 * 512);  // slot 0 = h0
    swz_gates_kernel<<<8192, 256, 0, stream>>>(Wf, Wc, Wi, Wo, wg);
    swz_wout_kernel<<<1024, 256, 0, stream>>>(Wout, wouts);

    lstm_persistent_kernel<<<32, 256, 0, stream>>>(c0, bf_, bc_, bi_, bo_,
                                                   xbf, hsb, wg, out_h, out_c, bar);

    outproj_kernel<<<2048, 256, 0, stream>>>(hsb, wouts, bout, out);
}
